// FlowEmbeddingLayer_51170240364939
// MI455X (gfx1250) — compile-verified, matrix operand order checked by reading, not run on hardware
//
#include <hip/hip_runtime.h>

#define LEAK 0.1f
#define BATCH 2
#define NPTS  8192
#define CCH   128
#define DCH   128
#define KNN   16

typedef __attribute__((ext_vector_type(2)))  float        v2f;
typedef __attribute__((ext_vector_type(8)))  float        v8f;
typedef __attribute__((ext_vector_type(4)))  unsigned int u32x4;
typedef __attribute__((ext_vector_type(8)))  unsigned int v8u;
typedef __attribute__((ext_vector_type(8)))  int          i32x8;
typedef __attribute__((ext_vector_type(4)))  int          i32x4;
typedef __attribute__((ext_vector_type(16))) __bf16       v16bf;

// ---- CDNA5 async / TDM feature detection (compile-safe fallbacks) ---------
#if defined(__HIP_DEVICE_COMPILE__) && __has_builtin(__builtin_amdgcn_global_load_async_to_lds_b128) && __has_builtin(__builtin_amdgcn_s_wait_asynccnt)
#define HAVE_ASYNC 1
#else
#define HAVE_ASYNC 0
#endif

#if defined(__HIP_DEVICE_COMPILE__) && __has_builtin(__builtin_amdgcn_tensor_load_to_lds) && __has_builtin(__builtin_amdgcn_s_wait_tensorcnt)
#define HAVE_TDM 1
#else
#define HAVE_TDM 0
#endif

__device__ __forceinline__ float leaky(float x) { return x > 0.f ? x : LEAK * x; }

// round-to-nearest-even f32 -> bf16
__device__ __forceinline__ unsigned short f2bf(float f) {
    unsigned int u = __builtin_bit_cast(unsigned int, f);
    u += 0x7FFFu + ((u >> 16) & 1u);
    return (unsigned short)(u >> 16);
}

#if HAVE_TDM || HAVE_ASYNC
__device__ __forceinline__ unsigned int lds_byte_off(const void* p) {
    return (unsigned int)(unsigned long long)
        (__attribute__((address_space(3))) const void*)p;
}
#endif

#if HAVE_TDM
// 1-D TDM copy: n_dwords f32 elements, global -> LDS (one D#, groups 2/3 NULL-ish)
__device__ __forceinline__ void tdm_load_1d(const void* gsrc, const void* ldst,
                                            unsigned int n_dwords) {
    unsigned long long ga = (unsigned long long)gsrc;
    u32x4 g0;
    g0.x = 1u;                                      // count=1, user descriptor
    g0.y = lds_byte_off(ldst);                      // lds_addr (bytes)
    g0.z = (unsigned int)ga;                        // global_addr[31:0]
    g0.w = (unsigned int)(ga >> 32) | (2u << 30);   // global_addr[56:32] | type=2
    i32x8 g1;
    g1[0] = (int)(2u << 16);                        // data_size = 4B
    g1[1] = (int)(n_dwords << 16);                  // tensor_dim0 low16 (bits 63:48)
    g1[2] = (int)((n_dwords >> 16) | (1u << 16));   // tensor_dim0 hi16 | tensor_dim1=1
    g1[3] = (int)(n_dwords << 16);                  // tile_dim0 (bits 127:112)
    g1[4] = 1;                                      // tile_dim1 = 1
    g1[5] = (int)n_dwords;                          // tensor_dim0_stride low32
    g1[6] = 0;
    g1[7] = 0;
    i32x4 z4 = {0, 0, 0, 0};
#if defined(__clang_major__) && __clang_major__ >= 23
    i32x8 z8 = {0, 0, 0, 0, 0, 0, 0, 0};
    __builtin_amdgcn_tensor_load_to_lds(g0, g1, z4, z4, z8, 0);
#else
    __builtin_amdgcn_tensor_load_to_lds(g0, g1, z4, z4, 0);
#endif
}
#endif

// 16-byte global -> LDS chunk copy (async path when available)
__device__ __forceinline__ void copy16(void* ldst, const void* gsrc) {
#if HAVE_ASYNC
    __builtin_amdgcn_global_load_async_to_lds_b128(
        (__attribute__((address_space(1))) i32x4*)gsrc,
        (__attribute__((address_space(3))) i32x4*)ldst, 0, 0);
#else
    *(u32x4*)ldst = *(const u32x4*)gsrc;
#endif
}
__device__ __forceinline__ void copy_fence() {
#if HAVE_ASYNC
    __builtin_amdgcn_s_wait_asynccnt(0);
#endif
}

// ---------------------------------------------------------------------------
// Kernel A: pack W_mlp1/W_mlp2 ([D=128][C=128] f32, row = out-channel) into
// the wave32 bf16 B-fragment layout for v_wmma_f32_16x16x32_bf16.
// flat[((t*4 + kc)*32 + lane)*8 + v] = {bf16(W[d][K0]), bf16(W[d][K0+1])}
//   lane: m = lane%16 -> N (out channel in tile), h = lane/16
//   v<4 : K0 = kc*32 + 8h + 2v ; v>=4 : K0 = kc*32 + 16 + 8h + 2(v-4)
// ---------------------------------------------------------------------------
__global__ void pack_weights(const float* __restrict__ Wm1,
                             const float* __restrict__ Wm2,
                             unsigned int* __restrict__ wp1,
                             unsigned int* __restrict__ wp2) {
    int tid = blockIdx.x * blockDim.x + threadIdx.x;   // 0..16383
    int which = tid >> 13;
    int r  = tid & 8191;
    int v  = r & 7;
    int lane = (r >> 3) & 31;
    int kc = (r >> 8) & 3;
    int t  = r >> 10;
    int m = lane & 15, h = lane >> 4;
    int K0 = kc * 32 + ((v < 4) ? (8 * h + 2 * v) : (16 + 8 * h + 2 * (v - 4)));
    int d  = t * 16 + m;
    const float* W = which ? Wm2 : Wm1;
    unsigned int lo = f2bf(W[d * CCH + K0]);
    unsigned int hi = f2bf(W[d * CCH + K0 + 1]);
    (which ? wp2 : wp1)[r] = lo | (hi << 16);
}

// ---------------------------------------------------------------------------
// Kernel B: p[b][n][d] = sum_c W[d][c] * points[b][c][n] + bias[d]
// Full-precision v_wmma_f32_16x16x4_f32: one 16(n)x16(d) tile per wave,
// K loop over C=128 in steps of 4 (32 WMMAs / tile).
// ---------------------------------------------------------------------------
__global__ void __launch_bounds__(128)
gemm_points(const float* __restrict__ pts1, const float* __restrict__ pts2,
            const float* __restrict__ Wt1,  const float* __restrict__ bt1,
            const float* __restrict__ Wt2,  const float* __restrict__ bt2,
            float* __restrict__ p1, float* __restrict__ p2) {
    int w = blockIdx.x * 4 + (threadIdx.x >> 5);
    int l = threadIdx.x & 31;
    int which = w >> 13;           // 0: points1, 1: points2
    int r  = w & 8191;
    int b  = r >> 12;
    int r2 = r & 4095;
    int n0 = (r2 >> 3) * 16;
    int d0 = (r2 & 7) * 16;
    int m = l & 15, h = l >> 4;

    const float* P    = which ? pts2 : pts1;
    const float* W    = which ? Wt2  : Wt1;
    const float* bias = which ? bt2  : bt1;
    float*       outp = which ? p2   : p1;

    const float* Pb = P + (size_t)b * CCH * NPTS;
    v8f acc = {0.f, 0.f, 0.f, 0.f, 0.f, 0.f, 0.f, 0.f};

#pragma unroll 4
    for (int c0 = 0; c0 < CCH; c0 += 4) {
        int ca = c0 + 2 * h;
        v2f a, bb;
        a.x  = Pb[(size_t)ca * NPTS + n0 + m];
        a.y  = Pb[(size_t)(ca + 1) * NPTS + n0 + m];
        bb.x = W[(d0 + m) * CCH + ca];
        bb.y = W[(d0 + m) * CCH + ca + 1];
        acc = __builtin_amdgcn_wmma_f32_16x16x4_f32(false, a, false, bb,
                                                    (short)0, acc, false, false);
    }
    float bv = bias[d0 + m];
#pragma unroll
    for (int v = 0; v < 8; ++v) {
        int n = n0 + v + 8 * h;
        outp[((size_t)b * NPTS + n) * DCH + d0 + m] = acc[v] + bv;
    }
}

// ---------------------------------------------------------------------------
// Kernel C: brute-force KNN (top-16 smallest squared distances).
// One wave per (b, n1), 4 waves/block (all same batch). Candidates are
// staged per-block into LDS in 1024-point chunks with async load-to-LDS
// (GLOBAL_LOAD_ASYNC_TO_LDS_B128 + s_wait_asynccnt), cutting vmem traffic 4x.
// Per-lane top-16 lists live in LDS (lane-private, conflict-free); lists are
// merged with 16 rounds of a shuffle-butterfly arg-min.
// ---------------------------------------------------------------------------
__global__ void __launch_bounds__(128)
knn_topk(const float* __restrict__ xyz1, const float* __restrict__ xyz2,
         int* __restrict__ idxout) {
    __shared__ __align__(16) float sPts[3 * 1024];   // x | y | z segments
    __shared__ float candD[4][KNN * 32];
    __shared__ int   candI[4][KNN * 32];

    int tid = threadIdx.x;
    int wv = tid >> 5;
    int l  = tid & 31;
    int p  = blockIdx.x * 4 + wv;
    int b  = p >> 13;                // all 4 waves in a block share b
    int n1 = p & 8191;

    const float* xq = xyz2 + (size_t)b * 3 * NPTS;
    const float* xp = xyz1 + (size_t)b * 3 * NPTS;
    float x1x = xp[n1], x1y = xp[NPTS + n1], x1z = xp[2 * NPTS + n1];

    float* cd = candD[wv];
    int*   ci = candI[wv];
#pragma unroll
    for (int s = 0; s < KNN; ++s) cd[s * 32 + l] = __builtin_inff();
    float cmax = __builtin_inff();
    int   cslot = 0;

    for (int c0 = 0; c0 < NPTS; c0 += 1024) {
        __syncthreads();             // previous chunk fully consumed
        // stage xyz2[:, c0:c0+1024] -> LDS: 768 x 16B chunks, 128 threads
        for (int i = tid; i < 768; i += 128) {
            int seg = i >> 8;            // 0=x, 1=y, 2=z
            int off = (i & 255) * 4;     // dword offset in segment
            copy16(&sPts[seg * 1024 + off], xq + (size_t)seg * NPTS + c0 + off);
        }
        copy_fence();
        __syncthreads();
        __builtin_prefetch(xq + c0 + 1024, 0, 1);   // speculative next-chunk hint

        for (int j0 = 0; j0 < 1024; j0 += 32) {
            int jj = j0 + l;
            float dx = sPts[jj] - x1x;
            float dy = sPts[1024 + jj] - x1y;
            float dz = sPts[2048 + jj] - x1z;
            float d = dx * dx + dy * dy + dz * dz;
            if (d < cmax) {              // replace current worst, rescan
                cd[cslot * 32 + l] = d;
                ci[cslot * 32 + l] = c0 + jj;
                cmax = -1.f;
#pragma unroll
                for (int s = 0; s < KNN; ++s) {
                    float v = cd[s * 32 + l];
                    if (v > cmax) { cmax = v; cslot = s; }
                }
            }
        }
    }

    // merge: 16 rounds of wave-wide arg-min over each lane's unconsumed slots
    unsigned int consumed = 0;
    for (int k = 0; k < KNN; ++k) {
        float bd = __builtin_inff();
        int   bc = 0;                    // (slot << 16) | idx   (idx < 8192)
#pragma unroll
        for (int s = 0; s < KNN; ++s) {
            if (!((consumed >> s) & 1u)) {
                float v = cd[s * 32 + l];
                if (v < bd) { bd = v; bc = (s << 16) | ci[s * 32 + l]; }
            }
        }
        int bl = l;
#pragma unroll
        for (int md = 16; md >= 1; md >>= 1) {
            float od = __shfl_xor(bd, md, 32);
            int   oc = __shfl_xor(bc, md, 32);
            int   ol = __shfl_xor(bl, md, 32);
            if (od < bd || (od == bd && ol < bl)) { bd = od; bc = oc; bl = ol; }
        }
        if (l == bl) consumed |= 1u << (bc >> 16);
        if (l == 0)  idxout[((size_t)b * NPTS + n1) * KNN + k] = bc & 0xFFFF;
    }
}

// ---------------------------------------------------------------------------
// Kernel D: per point: f0 = leaky(g2 + p1 + Wpos*dir + bpos) [16 x 128],
// two 16x128x128 bf16-WMMA MLP layers, max over the 16 neighbors.
// One wave per point, 4 waves/workgroup. W_pos/b_pos staged via the Tensor
// Data Mover (tensor_load_to_lds + s_wait_tensorcnt) issued by wave 0 and
// overlapped with the gather/direction setup of the other waves.
// Activations staged in LDS with a 136-element pitch (conflict-free gathers).
// ---------------------------------------------------------------------------
__global__ void __launch_bounds__(128)
embed_mlp(const float* __restrict__ xyz1, const float* __restrict__ xyz2,
          const float* __restrict__ p1,   const float* __restrict__ p2,
          const int* __restrict__ idxbuf,
          const float* __restrict__ Wpos, const float* __restrict__ bpos,
          const unsigned int* __restrict__ wp1, const float* __restrict__ b1,
          const unsigned int* __restrict__ wp2, const float* __restrict__ b2,
          float* __restrict__ dout) {
    __shared__ __align__(16) float sWpos[DCH * 3];
    __shared__ __align__(16) float sBpos[DCH];
    __shared__ unsigned short sF0[4][16 * 136];
    __shared__ unsigned short sF1[4][16 * 136];
    __shared__ float          sDir[4][KNN * 3];
    __shared__ int            sIdx[4][KNN];

    int tid = threadIdx.x;
    int wv = tid >> 5, l = tid & 31, m = l & 15, h = l >> 4;
    int pt = blockIdx.x * 4 + wv;
    int b  = pt >> 13, n1 = pt & 8191;

#if HAVE_TDM
    if (wv == 0) {                       // wave-level DMA, EXEC-independent
        tdm_load_1d(Wpos, sWpos, DCH * 3);
        tdm_load_1d(bpos, sBpos, DCH);
    }
#else
    for (int i = tid; i < DCH * 3; i += 128) sWpos[i] = Wpos[i];
    if (tid < DCH) sBpos[tid] = bpos[tid];
#endif

    if (l < KNN) {
        int j = idxbuf[((size_t)b * NPTS + n1) * KNN + l];
        sIdx[wv][l] = j;
        const float* x2 = xyz2 + (size_t)b * 3 * NPTS;
        const float* x1 = xyz1 + (size_t)b * 3 * NPTS;
        sDir[wv][l * 3 + 0] = x2[j] - x1[n1];
        sDir[wv][l * 3 + 1] = x2[NPTS + j] - x1[NPTS + n1];
        sDir[wv][l * 3 + 2] = x2[2 * NPTS + j] - x1[2 * NPTS + n1];
    }
#if HAVE_TDM
    if (wv == 0) __builtin_amdgcn_s_wait_tensorcnt(0);
#endif
    __syncthreads();

    // f0[k][d] = leaky(g2 + p1 + pos), stored bf16 in LDS
    const float* p1row = p1 + ((size_t)b * NPTS + n1) * DCH;
    const float* p2b   = p2 + (size_t)b * NPTS * DCH;
    for (int i = l; i < KNN * DCH; i += 32) {
        int k = i >> 7, d = i & 127;
        float dx = sDir[wv][k * 3], dy = sDir[wv][k * 3 + 1], dz = sDir[wv][k * 3 + 2];
        float pos = sBpos[d] + sWpos[d * 3] * dx + sWpos[d * 3 + 1] * dy + sWpos[d * 3 + 2] * dz;
        float val = p2b[(size_t)sIdx[wv][k] * DCH + d] + p1row[d] + pos;
        sF0[wv][k * 136 + d] = f2bf(leaky(val));
    }
    __syncthreads();

    // A-fragments for layer 1 (16-bit A 16x32 layout, K pairs per VGPR)
    const unsigned int* f0u = (const unsigned int*)sF0[wv];
    const unsigned int* f1u = (const unsigned int*)sF1[wv];
    v8u afrag[4];
#pragma unroll
    for (int kc = 0; kc < 4; ++kc)
#pragma unroll
        for (int v = 0; v < 8; ++v) {
            int K = kc * 32 + ((v < 4) ? (8 * h + 2 * v) : (16 + 8 * h + 2 * (v - 4)));
            afrag[kc][v] = f0u[(m * 136 + K) >> 1];
        }

    // MLP layer 1: 8 output tiles x 4 K-chunks of bf16 WMMA
#pragma unroll
    for (int t = 0; t < 8; ++t) {
        v8f acc = {0.f, 0.f, 0.f, 0.f, 0.f, 0.f, 0.f, 0.f};
#pragma unroll
        for (int kc = 0; kc < 4; ++kc) {
            v8u wb = *(const v8u*)(wp1 + ((size_t)(t * 4 + kc) * 32 + l) * 8);
            acc = __builtin_amdgcn_wmma_f32_16x16x32_bf16(
                false, __builtin_bit_cast(v16bf, afrag[kc]),
                false, __builtin_bit_cast(v16bf, wb),
                (short)0, acc, false, false);
        }
        float bv = b1[t * 16 + m];
#pragma unroll
        for (int v = 0; v < 8; ++v)
            sF1[wv][(v + 8 * h) * 136 + t * 16 + m] = f2bf(leaky(acc[v] + bv));
    }
    __syncthreads();

#pragma unroll
    for (int kc = 0; kc < 4; ++kc)
#pragma unroll
        for (int v = 0; v < 8; ++v) {
            int K = kc * 32 + ((v < 4) ? (8 * h + 2 * v) : (16 + 8 * h + 2 * (v - 4)));
            afrag[kc][v] = f1u[(m * 136 + K) >> 1];
        }

    // MLP layer 2 + max over the 16 neighbors (M dim), write [B, D, N]
#pragma unroll
    for (int t = 0; t < 8; ++t) {
        v8f acc = {0.f, 0.f, 0.f, 0.f, 0.f, 0.f, 0.f, 0.f};
#pragma unroll
        for (int kc = 0; kc < 4; ++kc) {
            v8u wb = *(const v8u*)(wp2 + ((size_t)(t * 4 + kc) * 32 + l) * 8);
            acc = __builtin_amdgcn_wmma_f32_16x16x32_bf16(
                false, __builtin_bit_cast(v16bf, afrag[kc]),
                false, __builtin_bit_cast(v16bf, wb),
                (short)0, acc, false, false);
        }
        float bv = b2[t * 16 + m];
        float mx = -__builtin_inff();
#pragma unroll
        for (int v = 0; v < 8; ++v) mx = fmaxf(mx, leaky(acc[v] + bv));
        mx = fmaxf(mx, __shfl_xor(mx, 16, 32));   // fold M=0..7 with M=8..15
        if (h == 0)
            dout[((size_t)b * DCH + t * 16 + m) * NPTS + n1] = mx;
    }
}

// ---------------------------------------------------------------------------
extern "C" void kernel_launch(void* const* d_in, const int* in_sizes, int n_in,
                              void* d_out, int out_size, void* d_ws, size_t ws_size,
                              hipStream_t stream) {
    (void)in_sizes; (void)n_in; (void)out_size; (void)ws_size;
    const float* xyz1    = (const float*)d_in[0];
    const float* xyz2    = (const float*)d_in[1];
    const float* points1 = (const float*)d_in[2];
    const float* points2 = (const float*)d_in[3];
    const float* Wt11    = (const float*)d_in[4];
    const float* bt11    = (const float*)d_in[5];
    const float* Wt22    = (const float*)d_in[6];
    const float* bt22    = (const float*)d_in[7];
    const float* Wpos    = (const float*)d_in[8];
    const float* bpos    = (const float*)d_in[9];
    const float* Wm1     = (const float*)d_in[10];
    const float* bm1     = (const float*)d_in[11];
    const float* Wm2     = (const float*)d_in[12];
    const float* bm2     = (const float*)d_in[13];
    float* out = (float*)d_out;

    // workspace layout (f32 elements): p1 | p2 | idx | wpack1 | wpack2  (~17.1 MB)
    float* ws = (float*)d_ws;
    float* p1 = ws;
    float* p2 = ws + 2097152;
    int*   idx = (int*)(ws + 4194304);
    unsigned int* wp1 = (unsigned int*)(ws + 4456448);
    unsigned int* wp2 = (unsigned int*)(ws + 4464640);

    pack_weights<<<64, 256, 0, stream>>>(Wm1, Wm2, wp1, wp2);
    gemm_points<<<4096, 128, 0, stream>>>(points1, points2, Wt11, bt11, Wt22, bt22, p1, p2);
    knn_topk<<<4096, 128, 0, stream>>>(xyz1, xyz2, idx);
    embed_mlp<<<4096, 128, 0, stream>>>(xyz1, xyz2, p1, p2, idx,
                                        Wpos, bpos, wp1, bm1, wp2, bm2, out);
}
